// HangulViT_5214090297787
// MI455X (gfx1250) — matrix operand order repaired and checked by reading, not compiled, not run on hardware
//
#include <hip/hip_runtime.h>
#include <hip/hip_bf16.h>

typedef unsigned short u16;
typedef __bf16 v16bf __attribute__((ext_vector_type(16)));
typedef float  v8f   __attribute__((ext_vector_type(8)));

#define F_BIAS  1
#define F_RELU  2
#define F_BF    4
#define F_CT    8   // store C transposed: C[n*ldc + m] (used to build V^T)

__device__ __forceinline__ u16 f2bf(float f) {
    unsigned u = __float_as_uint(f);
    unsigned r = (u + 0x7FFFu + ((u >> 16) & 1u)) >> 16;
    return (u16)r;
}

__device__ __forceinline__ v16bf ld16(const u16* p0, const u16* p1) {
    union { uint4 u[2]; v16bf v; } r;
    r.u[0] = *(const uint4*)p0;
    r.u[1] = *(const uint4*)p1;
    return r.v;
}

// CDNA5 async copy: 16 bytes global -> LDS, tracked by ASYNCcnt.
// Low 32 bits of a generic pointer to LDS are the LDS byte offset.
__device__ __forceinline__ void async_b128(const u16* g, u16* l) {
    unsigned loff = (unsigned)(unsigned long long)l;
    asm volatile("global_load_async_to_lds_b128 %0, %1, off"
                 :: "v"(loff), "v"(g) : "memory");
}
__device__ __forceinline__ void wait_async0() {
    asm volatile("s_wait_asynccnt 0x0" ::: "memory");
}

// ---------------------------------------------------------------------------
// Batched bf16 GEMM, fp32 accumulate via V_WMMA_F32_16X16X32_BF16.
// 256 threads = 8 waves; tile 128(M) x 64(N); K-step 32; double-buffered LDS
// filled exclusively with global_load_async_to_lds_b128 (branch-free, clamped
// addressing: clamped A-rows / B-cols only feed outputs the store guards drop).
//
// A: bf16 [M,K] row-major (lda).
// B: bf16, k-contiguous: element(k,n) at B[n*ldb + k]  (weights are staged
//    transposed; K-matrices and V^T are naturally k-contiguous).
// Batch z decomposed as (zb,zh) = (z/nH, z%nH) with independent strides.
// All lda/ldb/stride/k offsets are multiples of 8 elements (16B alignment).
// ---------------------------------------------------------------------------
template<int F>
__global__ __launch_bounds__(256)
void gemm_bf16_kernel(const u16* __restrict__ A, const u16* __restrict__ B,
                      void* __restrict__ Cv, const float* __restrict__ bias,
                      int M, int N, int K,
                      int lda, int ldb, int ldc, int nH,
                      long sAb, long sAh, long sBb, long sBh, long sCb, long sCh,
                      float alpha)
{
    __shared__ __align__(16) u16 As[2][128 * 32];
    __shared__ __align__(16) u16 Bs[2][64 * 32];    // n-major: Bs[n][k]
    const int tid = threadIdx.x;
    const int z  = blockIdx.z;
    const int zb = z / nH, zh = z % nH;
    A += (long)zb * sAb + (long)zh * sAh;
    B += (long)zb * sBb + (long)zh * sBh;
    const long coff = (long)zb * sCb + (long)zh * sCh;
    const int tileM = blockIdx.y * 128;
    const int tileN = blockIdx.x * 64;
    const int wave = tid >> 5, lane = tid & 31;
    const int wm = wave >> 1, wn = wave & 1;
    const int laneLo = lane & 15;
    const int kbA = (lane >> 4) * 8;    // A frag: K halves 0..7 / 8..15 (+16)
    const int kbB = (lane >> 4) * 8;    // B frag: K halves 0..7 / 8..15 (+16)

    // staging addresses for this thread (branch-free, clamped)
    const int rA0 = ((tid * 8) >> 5),      cA0 = (tid * 8) & 31;
    const int rA1 = (((tid + 256) * 8) >> 5), cA1 = ((tid + 256) * 8) & 31;
    const int grA0 = min(tileM + rA0, M - 1);
    const int grA1 = min(tileM + rA1, M - 1);
    const int nB = tid >> 2, cB = (tid & 3) * 8;
    const int gnB = min(tileN + nB, N - 1);

    v8f acc[2][2];
#pragma unroll
    for (int a = 0; a < 2; ++a)
#pragma unroll
        for (int b = 0; b < 2; ++b)
#pragma unroll
            for (int i = 0; i < 8; ++i) acc[a][b][i] = 0.f;

    // prologue: stage k0 = 0 into buffer 0
    async_b128(A + (long)grA0 * lda + cA0, &As[0][rA0 * 32 + cA0]);
    async_b128(A + (long)grA1 * lda + cA1, &As[0][rA1 * 32 + cA1]);
    async_b128(B + (long)gnB * ldb + cB,  &Bs[0][nB * 32 + cB]);
    wait_async0();
    __syncthreads();

    int buf = 0;
    for (int k0 = 0; k0 < K; k0 += 32) {
        const bool more = (k0 + 32) < K;
        if (more) {   // issue async copies for next k-slab into alt buffer
            int kn = k0 + 32;
            async_b128(A + (long)grA0 * lda + kn + cA0, &As[buf ^ 1][rA0 * 32 + cA0]);
            async_b128(A + (long)grA1 * lda + kn + cA1, &As[buf ^ 1][rA1 * 32 + cA1]);
            async_b128(B + (long)gnB * ldb + kn + cB,  &Bs[buf ^ 1][nB * 32 + cB]);
        }

        const u16* as = &As[buf][0];
        const u16* bs = &Bs[buf][0];
        v16bf af[2], bfrag[2];
#pragma unroll
        for (int mt = 0; mt < 2; ++mt) {
            int r = wm * 32 + mt * 16 + laneLo;
            af[mt] = ld16(&as[r * 32 + kbA], &as[r * 32 + kbA + 16]);
        }
#pragma unroll
        for (int nt = 0; nt < 2; ++nt) {
            int c = wn * 32 + nt * 16 + laneLo;
            bfrag[nt] = ld16(&bs[c * 32 + kbB], &bs[c * 32 + kbB + 16]);
        }
#pragma unroll
        for (int mt = 0; mt < 2; ++mt)
#pragma unroll
            for (int nt = 0; nt < 2; ++nt)
                acc[mt][nt] = __builtin_amdgcn_wmma_f32_16x16x32_bf16(
                    false, af[mt], false, bfrag[nt], (short)0, acc[mt][nt],
                    false, false);

        if (more) wait_async0();
        __syncthreads();
        buf ^= 1;
    }

    float* Cf = (float*)Cv;
    u16*   Cb = (u16*)Cv;
    const int rb = (lane >> 4) * 8;   // C frag: rows 0..7 / 8..15 per half-wave
#pragma unroll
    for (int mt = 0; mt < 2; ++mt) {
#pragma unroll
        for (int nt = 0; nt < 2; ++nt) {
            int gn = tileN + wn * 32 + nt * 16 + laneLo;
            if (gn >= N) continue;
            float bv = (F & F_BIAS) ? bias[gn] : 0.f;
#pragma unroll
            for (int r = 0; r < 8; ++r) {
                int gm = tileM + wm * 32 + mt * 16 + rb + r;
                if (gm < M) {
                    float v = acc[mt][nt][r] * alpha + bv;
                    if (F & F_RELU) v = fmaxf(v, 0.f);
                    long idx = (F & F_CT) ? (coff + (long)gn * ldc + gm)
                                          : (coff + (long)gm * ldc + gn);
                    if (F & F_BF) Cb[idx] = f2bf(v);
                    else          Cf[idx] = v;
                }
            }
        }
    }
}

// ---------------------------------------------------------------------------
// Elementwise / reduction kernels
// ---------------------------------------------------------------------------
// f32 [K,N] row-major -> bf16 [N,K] (k-contiguous weight staging)
__global__ __launch_bounds__(256)
void convT_kernel(const float* __restrict__ s, u16* __restrict__ d, int K, int N)
{
    long i = (long)blockIdx.x * 256 + threadIdx.x;
    if (i >= (long)K * N) return;
    int k = (int)(i / N), n = (int)(i % N);
    d[(long)n * K + k] = f2bf(s[i]);
}

__global__ __launch_bounds__(256)
void patchify_kernel(const float* __restrict__ img, u16* __restrict__ out)
{
    long i = (long)blockIdx.x * 256 + threadIdx.x;
    if (i >= (long)15488 * 768) return;
    int m = (int)(i / 768), c = (int)(i % 768);
    int b = m / 121, f = m % 121;
    int fy = f / 11, fx = f % 11;
    int ch = c / 256, rem = c % 256;
    int py = rem / 16, px = rem % 16;
    float v = img[(((long)b * 3 + ch) * 176 + fy * 16 + py) * 176 + fx * 16 + px];
    out[i] = f2bf(v);
}

__global__ __launch_bounds__(256)
void addpos_kernel(float* __restrict__ xf, u16* __restrict__ xb,
                   const float* __restrict__ pos)
{
    long i = (long)blockIdx.x * 256 + threadIdx.x;
    if (i >= (long)15488 * 512) return;
    int m = (int)(i / 512), d = (int)(i % 512);
    int t = m % 121;
    float v = xf[i] + pos[t * 512 + d];
    xf[i] = v;
    xb[i] = f2bf(v);
}

__global__ __launch_bounds__(256)
void embed_kernel(const int* __restrict__ labels, const float* __restrict__ emb,
                  const float* __restrict__ pos, float* __restrict__ yf,
                  u16* __restrict__ yb)
{
    long i = (long)blockIdx.x * 256 + threadIdx.x;
    if (i >= (long)4096 * 512) return;
    int m = (int)(i / 512), d = (int)(i % 512);
    int t = m % 32;
    int lab = labels[m];
    float v = emb[(long)lab * 512 + d] * 22.62741699796952f + pos[t * 512 + d];
    yf[i] = v;
    yb[i] = f2bf(v);
}

// fused: out = LN(resid + branch) * gamma + beta ; writes f32 + bf16 copies
__global__ __launch_bounds__(256)
void ln_fused_kernel(const float* __restrict__ resid, const float* __restrict__ branch,
                     const float* __restrict__ gamma, const float* __restrict__ beta,
                     float* __restrict__ outf, u16* __restrict__ outb)
{
    __shared__ float sm[16];
    int row = blockIdx.x, t = threadIdx.x;
    const float* r  = resid  + (long)row * 512;
    const float* br = branch + (long)row * 512;
    float a  = r[t]       + br[t];
    float b2 = r[t + 256] + br[t + 256];
    float s = a + b2, ss = a * a + b2 * b2;
    for (int off = 16; off; off >>= 1) {
        s  += __shfl_xor(s,  off, 32);
        ss += __shfl_xor(ss, off, 32);
    }
    int wave = t >> 5, lane = t & 31;
    if (lane == 0) { sm[wave] = s; sm[8 + wave] = ss; }
    __syncthreads();
    if (t == 0) {
        float ts = 0.f, tss = 0.f;
        for (int i = 0; i < 8; ++i) { ts += sm[i]; tss += sm[8 + i]; }
        sm[0] = ts; sm[1] = tss;
    }
    __syncthreads();
    float mean = sm[0] * (1.f / 512.f);
    float var  = sm[1] * (1.f / 512.f) - mean * mean;
    float rstd = rsqrtf(var + 1e-5f);
    float y0 = (a  - mean) * rstd * gamma[t]       + beta[t];
    float y1 = (b2 - mean) * rstd * gamma[t + 256] + beta[t + 256];
    outf[(long)row * 512 + t]       = y0;
    outf[(long)row * 512 + t + 256] = y1;
    outb[(long)row * 512 + t]       = f2bf(y0);
    outb[(long)row * 512 + t + 256] = f2bf(y1);
}

// encoder attention softmax: rows (b,h,q) over 128 cols (121 valid), fp32 in,
// bf16 probabilities written IN PLACE (row stride stays 128 f32 = 256 u16);
// padded columns written as exact zeros so K-padding of P*V contributes 0.
__global__ __launch_bounds__(256)
void softmax_enc_kernel(float* __restrict__ S)
{
    int gid = blockIdx.x * 8 + (threadIdx.x >> 5);
    int lane = threadIdx.x & 31;
    int q = gid & 127, bh = gid >> 7;
    if (bh >= 1024 || q >= 121) return;
    float* row = S + ((long)bh * 128 + q) * 128;
    float v[4], e[4];
#pragma unroll
    for (int j = 0; j < 4; ++j) {
        int c = lane + j * 32;
        v[j] = (c < 121) ? row[c] : -3.0e38f;
    }
    float m = fmaxf(fmaxf(v[0], v[1]), fmaxf(v[2], v[3]));
    for (int off = 16; off; off >>= 1) m = fmaxf(m, __shfl_xor(m, off, 32));
    float s = 0.f;
#pragma unroll
    for (int j = 0; j < 4; ++j) {
        int c = lane + j * 32;
        e[j] = (c < 121) ? __expf(v[j] - m) : 0.f;
        s += e[j];
    }
    for (int off = 16; off; off >>= 1) s += __shfl_xor(s, off, 32);
    float inv = 1.f / s;
    u16* prow = (u16*)row;
#pragma unroll
    for (int j = 0; j < 4; ++j) prow[lane + j * 32] = f2bf(e[j] * inv);
}

// decoder self-attn softmax: 32 cols, causal + pad mask from labels (additive -1e9)
__global__ __launch_bounds__(256)
void softmax_dec_self_kernel(float* __restrict__ S, const int* __restrict__ labels)
{
    int gid = blockIdx.x * 8 + (threadIdx.x >> 5);
    int lane = threadIdx.x & 31;
    if (gid >= 32768) return;
    int q = gid & 31, bh = gid >> 5, b = bh >> 3;
    float* row = S + ((long)bh * 32 + q) * 32;
    bool np = labels[b * 32 + lane] != 53;
    unsigned ok = (unsigned)__ballot(np);
    unsigned pref = (q == 31) ? 0xFFFFFFFFu : ((1u << (q + 1)) - 1u);
    bool rowok = (ok & pref) == pref;
    bool keep = rowok && np && (lane <= q);
    float x = row[lane] + (keep ? 0.f : -1e9f);
    float m = x;
    for (int off = 16; off; off >>= 1) m = fmaxf(m, __shfl_xor(m, off, 32));
    float e = __expf(x - m);
    float s = e;
    for (int off = 16; off; off >>= 1) s += __shfl_xor(s, off, 32);
    u16* prow = (u16*)row;
    prow[lane] = f2bf(e / s);
}

// cross-attn softmax: 128 cols (121 valid), additive -1e9 when row_ok false
__global__ __launch_bounds__(256)
void softmax_cross_kernel(float* __restrict__ S, const int* __restrict__ labels)
{
    int gid = blockIdx.x * 8 + (threadIdx.x >> 5);
    int lane = threadIdx.x & 31;
    if (gid >= 32768) return;
    int q = gid & 31, bh = gid >> 5, b = bh >> 3;
    float* row = S + ((long)bh * 32 + q) * 128;
    bool np = labels[b * 32 + lane] != 53;
    unsigned ok = (unsigned)__ballot(np);
    unsigned pref = (q == 31) ? 0xFFFFFFFFu : ((1u << (q + 1)) - 1u);
    float addm = ((ok & pref) == pref) ? 0.f : -1e9f;
    float v[4], e[4];
#pragma unroll
    for (int j = 0; j < 4; ++j) {
        int c = lane + j * 32;
        v[j] = (c < 121) ? (row[c] + addm) : -3.0e38f;
    }
    float m = fmaxf(fmaxf(v[0], v[1]), fmaxf(v[2], v[3]));
    for (int off = 16; off; off >>= 1) m = fmaxf(m, __shfl_xor(m, off, 32));
    float s = 0.f;
#pragma unroll
    for (int j = 0; j < 4; ++j) {
        int c = lane + j * 32;
        e[j] = (c < 121) ? __expf(v[j] - m) : 0.f;
        s += e[j];
    }
    for (int off = 16; off; off >>= 1) s += __shfl_xor(s, off, 32);
    float inv = 1.f / s;
    u16* prow = (u16*)row;
#pragma unroll
    for (int j = 0; j < 4; ++j) prow[lane + j * 32] = f2bf(e[j] * inv);
}

__global__ __launch_bounds__(256)
void logsoftmax_kernel(const float* __restrict__ logits, float* __restrict__ out)
{
    int row = blockIdx.x * 8 + (threadIdx.x >> 5);
    int lane = threadIdx.x & 31;
    if (row >= 4096) return;
    const float* rp = logits + (long)row * 64;
    float a = (lane < 54)      ? rp[lane]      : -3.0e38f;
    float b = (lane + 32 < 54) ? rp[lane + 32] : -3.0e38f;
    float m = fmaxf(a, b);
    for (int off = 16; off; off >>= 1) m = fmaxf(m, __shfl_xor(m, off, 32));
    float s = ((lane < 54) ? __expf(a - m) : 0.f) +
              ((lane + 32 < 54) ? __expf(b - m) : 0.f);
    for (int off = 16; off; off >>= 1) s += __shfl_xor(s, off, 32);
    float ls = m + __logf(s);
    if (lane < 54)      out[(long)row * 54 + lane]      = a - ls;
    if (lane + 32 < 54) out[(long)row * 54 + lane + 32] = b - ls;
}

// ---------------------------------------------------------------------------
template<int F>
static inline void run_gemm(hipStream_t st, const u16* A, const u16* B, void* C,
                            const float* bias, int M, int N, int K,
                            int lda, int ldb, int ldc, int nH,
                            long sAb, long sAh, long sBb, long sBh,
                            long sCb, long sCh, int batches, float alpha)
{
    dim3 g((N + 63) / 64, (M + 127) / 128, batches);
    gemm_bf16_kernel<F><<<g, dim3(256), 0, st>>>(
        A, B, C, bias, M, N, K, lda, ldb, ldc, nH,
        sAb, sAh, sBb, sBh, sCb, sCh, alpha);
}

extern "C" void kernel_launch(void* const* d_in, const int* in_sizes, int n_in,
                              void* d_out, int out_size, void* d_ws, size_t ws_size,
                              hipStream_t stream)
{
    const float* image      = (const float*)d_in[0];
    const int*   labels     = (const int*)  d_in[1];
    const float* patch_w    = (const float*)d_in[2];
    const float* patch_b    = (const float*)d_in[3];
    const float* enc_pos    = (const float*)d_in[4];
    const float* tok_emb    = (const float*)d_in[5];
    const float* dec_pos    = (const float*)d_in[6];
    const float* enc_attn_w = (const float*)d_in[7];
    const float* enc_ffn_w1 = (const float*)d_in[8];
    const float* enc_ffn_w2 = (const float*)d_in[9];
    const float* enc_ln     = (const float*)d_in[10];
    const float* dec_self_w = (const float*)d_in[11];
    const float* dec_cross_w= (const float*)d_in[12];
    const float* dec_ffn_w1 = (const float*)d_in[13];
    const float* dec_ffn_w2 = (const float*)d_in[14];
    const float* dec_ln     = (const float*)d_in[15];
    const float* gen_w      = (const float*)d_in[16];
    const float* gen_b      = (const float*)d_in[17];

    char* base = (char*)d_ws;
    size_t off = 0;
    auto alloc = [&](size_t bytes) -> char* {
        char* p = base + off;
        off += (bytes + 255) & ~(size_t)255;
        return p;
    };
    float* x_f    = (float*)alloc((size_t)15488 * 512 * 4);
    u16*   x_b    = (u16*)  alloc((size_t)15488 * 512 * 2);   // enc_out (bf16) after loop
    float* t_f    = (float*)alloc((size_t)15488 * 512 * 4);
    u16*   q_b    = (u16*)  alloc((size_t)15488 * 512 * 2);
    u16*   k_b    = (u16*)  alloc((size_t)15488 * 512 * 2);
    u16*   vT     = (u16*)  alloc((size_t)512 * 16384 * 2);   // V^T: [512, B*128pad]
    u16*   at_b   = (u16*)  alloc((size_t)15488 * 512 * 2);
    u16*   h_b    = (u16*)  alloc((size_t)15488 * 2048 * 2);
    float* S      = (float*)alloc((size_t)1024 * 128 * 128 * 4);
    float* y_f    = (float*)alloc((size_t)4096 * 512 * 4);
    u16*   y_b    = (u16*)  alloc((size_t)4096 * 512 * 2);
    float* logits = (float*)alloc((size_t)4096 * 64 * 4);
    u16*   wst    = (u16*)  alloc((size_t)2048 * 512 * 2);    // transposed weight staging
    u16*   patches = (u16*)h_b;   // alias: patches dead once FFN runs

    auto convT = [&](const float* s, u16* d, int K, int N) {
        long n = (long)K * N;
        convT_kernel<<<dim3((unsigned)((n + 255) / 256)), dim3(256), 0, stream>>>(s, d, K, N);
    };

    // ---------------- encoder: patchify + embed ----------------
    patchify_kernel<<<dim3((15488u * 768u + 255) / 256), dim3(256), 0, stream>>>(image, patches);
    convT(patch_w, wst, 768, 512);
    run_gemm<F_BIAS>(stream, patches, wst, x_f, patch_b, 15488, 512, 768,
                     768, 768, 512, 1, 0, 0, 0, 0, 0, 0, 1, 1.f);
    addpos_kernel<<<dim3((15488u * 512u + 255) / 256), dim3(256), 0, stream>>>(x_f, x_b, enc_pos);

    for (int i = 0; i < 4; ++i) {
        const float* W = enc_attn_w + (size_t)i * 4 * 262144;
        convT(W + 0 * 262144, wst, 512, 512);
        run_gemm<F_BF>(stream, x_b, wst, q_b, nullptr, 15488, 512, 512,
                       512, 512, 512, 1, 0, 0, 0, 0, 0, 0, 1, 1.f);
        convT(W + 1 * 262144, wst, 512, 512);
        run_gemm<F_BF>(stream, x_b, wst, k_b, nullptr, 15488, 512, 512,
                       512, 512, 512, 1, 0, 0, 0, 0, 0, 0, 1, 1.f);
        // V projection, stored transposed into vT[dh_global][b*128 + t]
        convT(W + 2 * 262144, wst, 512, 512);
        run_gemm<F_BF | F_CT>(stream, x_b, wst, vT, nullptr, 121, 512, 512,
                              512, 512, 16384, 1, 61952, 0, 0, 0, 128, 0,
                              128, 1.f);
        // S[b,h] = Q K^T / 8   (batched over B*H = 1024; K-matrix is k-contiguous)
        run_gemm<0>(stream, q_b, k_b, S, nullptr, 121, 121, 64,
                    512, 512, 128, 8, 61952, 64, 61952, 64, 131072, 16384,
                    1024, 0.125f);
        softmax_enc_kernel<<<dim3(16384), dim3(256), 0, stream>>>(S);
        // O = P V   (P bf16 in-place in S rows: lda = 256 u16; B = V^T)
        run_gemm<F_BF>(stream, (const u16*)S, vT, at_b, nullptr, 121, 64, 128,
                       256, 16384, 512, 8, 262144, 32768, 128, 1048576,
                       61952, 64, 1024, 1.f);
        convT(W + 3 * 262144, wst, 512, 512);
        run_gemm<0>(stream, at_b, wst, t_f, nullptr, 15488, 512, 512,
                    512, 512, 512, 1, 0, 0, 0, 0, 0, 0, 1, 1.f);
        ln_fused_kernel<<<dim3(15488), dim3(256), 0, stream>>>(
            x_f, t_f, enc_ln + (size_t)(i * 4 + 0) * 512,
            enc_ln + (size_t)(i * 4 + 1) * 512, x_f, x_b);
        // FFN
        convT(enc_ffn_w1 + (size_t)i * 1048576, wst, 512, 2048);
        run_gemm<F_RELU | F_BF>(stream, x_b, wst, h_b, nullptr, 15488, 2048, 512,
                                512, 512, 2048, 1, 0, 0, 0, 0, 0, 0, 1, 1.f);
        convT(enc_ffn_w2 + (size_t)i * 1048576, wst, 2048, 512);
        run_gemm<0>(stream, h_b, wst, t_f, nullptr, 15488, 512, 2048,
                    2048, 2048, 512, 1, 0, 0, 0, 0, 0, 0, 1, 1.f);
        ln_fused_kernel<<<dim3(15488), dim3(256), 0, stream>>>(
            x_f, t_f, enc_ln + (size_t)(i * 4 + 2) * 512,
            enc_ln + (size_t)(i * 4 + 3) * 512, x_f, x_b);
    }
    // x_b now holds enc_out in bf16 (persists through decoder)

    // ---------------- decoder ----------------
    embed_kernel<<<dim3((4096u * 512u + 255) / 256), dim3(256), 0, stream>>>(
        labels, tok_emb, dec_pos, y_f, y_b);

    for (int i = 0; i < 4; ++i) {
        // self attention
        const float* Ws = dec_self_w + (size_t)i * 4 * 262144;
        convT(Ws + 0 * 262144, wst, 512, 512);
        run_gemm<F_BF>(stream, y_b, wst, q_b, nullptr, 4096, 512, 512,
                       512, 512, 512, 1, 0, 0, 0, 0, 0, 0, 1, 1.f);
        convT(Ws + 1 * 262144, wst, 512, 512);
        run_gemm<F_BF>(stream, y_b, wst, k_b, nullptr, 4096, 512, 512,
                       512, 512, 512, 1, 0, 0, 0, 0, 0, 0, 1, 1.f);
        convT(Ws + 2 * 262144, wst, 512, 512);
        run_gemm<F_BF | F_CT>(stream, y_b, wst, vT, nullptr, 4096, 512, 512,
                              512, 512, 4096, 1, 0, 0, 0, 0, 0, 0, 1, 1.f);
        run_gemm<0>(stream, q_b, k_b, S, nullptr, 32, 32, 64,
                    512, 512, 32, 8, 16384, 64, 16384, 64, 8192, 1024,
                    1024, 0.125f);
        softmax_dec_self_kernel<<<dim3(4096), dim3(256), 0, stream>>>(S, labels);
        run_gemm<F_BF>(stream, (const u16*)S, vT, at_b, nullptr, 32, 64, 32,
                       64, 4096, 512, 8, 16384, 2048, 32, 262144,
                       16384, 64, 1024, 1.f);
        convT(Ws + 3 * 262144, wst, 512, 512);
        run_gemm<0>(stream, at_b, wst, t_f, nullptr, 4096, 512, 512,
                    512, 512, 512, 1, 0, 0, 0, 0, 0, 0, 1, 1.f);
        ln_fused_kernel<<<dim3(4096), dim3(256), 0, stream>>>(
            y_f, t_f, dec_ln + (size_t)(i * 6 + 0) * 512,
            dec_ln + (size_t)(i * 6 + 1) * 512, y_f, y_b);

        // cross attention (K/V from enc_out = x_b)
        const float* Wc = dec_cross_w + (size_t)i * 4 * 262144;
        convT(Wc + 0 * 262144, wst, 512, 512);
        run_gemm<F_BF>(stream, y_b, wst, q_b, nullptr, 4096, 512, 512,
                       512, 512, 512, 1, 0, 0, 0, 0, 0, 0, 1, 1.f);
        convT(Wc + 1 * 262144, wst, 512, 512);
        run_gemm<F_BF>(stream, x_b, wst, k_b, nullptr, 15488, 512, 512,
                       512, 512, 512, 1, 0, 0, 0, 0, 0, 0, 1, 1.f);
        convT(Wc + 2 * 262144, wst, 512, 512);
        run_gemm<F_BF | F_CT>(stream, x_b, wst, vT, nullptr, 121, 512, 512,
                              512, 512, 16384, 1, 61952, 0, 0, 0, 128, 0,
                              128, 1.f);
        run_gemm<0>(stream, q_b, k_b, S, nullptr, 32, 121, 64,
                    512, 512, 128, 8, 16384, 64, 61952, 64, 32768, 4096,
                    1024, 0.125f);
        softmax_cross_kernel<<<dim3(4096), dim3(256), 0, stream>>>(S, labels);
        run_gemm<F_BF>(stream, (const u16*)S, vT, at_b, nullptr, 32, 64, 128,
                       256, 16384, 512, 8, 65536, 8192, 128, 1048576,
                       16384, 64, 1024, 1.f);
        convT(Wc + 3 * 262144, wst, 512, 512);
        run_gemm<0>(stream, at_b, wst, t_f, nullptr, 4096, 512, 512,
                    512, 512, 512, 1, 0, 0, 0, 0, 0, 0, 1, 1.f);
        ln_fused_kernel<<<dim3(4096), dim3(256), 0, stream>>>(
            y_f, t_f, dec_ln + (size_t)(i * 6 + 2) * 512,
            dec_ln + (size_t)(i * 6 + 3) * 512, y_f, y_b);

        // FFN
        convT(dec_ffn_w1 + (size_t)i * 1048576, wst, 512, 2048);
        run_gemm<F_RELU | F_BF>(stream, y_b, wst, h_b, nullptr, 4096, 2048, 512,
                                512, 512, 2048, 1, 0, 0, 0, 0, 0, 0, 1, 1.f);
        convT(dec_ffn_w2 + (size_t)i * 1048576, wst, 2048, 512);
        run_gemm<0>(stream, h_b, wst, t_f, nullptr, 4096, 512, 2048,
                    2048, 2048, 512, 1, 0, 0, 0, 0, 0, 0, 1, 1.f);
        ln_fused_kernel<<<dim3(4096), dim3(256), 0, stream>>>(
            y_f, t_f, dec_ln + (size_t)(i * 6 + 4) * 512,
            dec_ln + (size_t)(i * 6 + 5) * 512, y_f, y_b);
    }

    // ---------------- generator + log_softmax ----------------
    convT(gen_w, wst, 512, 54);
    run_gemm<F_BIAS>(stream, y_b, wst, logits, gen_b, 4096, 54, 512,
                     512, 512, 64, 1, 0, 0, 0, 0, 0, 0, 1, 1.f);
    logsoftmax_kernel<<<dim3(512), dim3(256), 0, stream>>>(logits, (float*)d_out);

    (void)in_sizes; (void)n_in; (void)out_size; (void)ws_size;
}
